// CausalSelfAttention_90288802497024
// MI455X (gfx1250) — matrix operand order repaired and checked by reading, not compiled, not run on hardware
//
#include <hip/hip_runtime.h>

// ---------------------------------------------------------------------------
// Causal self-attention for MI455X (gfx1250, wave32, WMMA bf16 16x16x32).
// All GEMMs (QKV, QK^T, PV, proj) run on v_wmma_f32_16x16x32_bf16 with f32
// accumulation. bf16 intermediates: Q,K,V^T (48MB) fit in the 192MB L2 ->
// attention's O(T^2) traffic is L2-resident. GEMMs are software-pipelined
// with explicit ping-pong register sets (no rotation copies) so WMMAs overlap
// L2 loads with partial s_wait_loadcnt.
// ---------------------------------------------------------------------------

#define B_ 4
#define T_ 2048
#define C_ 1024
#define H_ 16
#define D_ 64
#define SCALE_ 0.125f   // D^-0.5

typedef __attribute__((ext_vector_type(16))) __bf16 v16bf;
typedef __attribute__((ext_vector_type(8)))  float  v8f;
typedef __attribute__((ext_vector_type(4)))  unsigned int v4u;

union V16U { v4u q[2]; v16bf v; };

static __device__ __forceinline__ unsigned short f2bf(float f) {
  unsigned int u = __builtin_bit_cast(unsigned int, f);
  unsigned int r = u + 0x7FFFu + ((u >> 16) & 1u);   // round-to-nearest-even
  return (unsigned short)(r >> 16);
}

// A-operand (16x32 bf16, row-major source): lane holds row m = lane%16.
// VGPR0-3: K = 8*hf .. 8*hf+7 ; VGPR4-7: K = 16+8*hf .. 16+8*hf+7.
static __device__ __forceinline__ v16bf load_a16(const unsigned short* row, int kc, int hf) {
  V16U t;
  t.q[0] = *(const v4u*)(row + kc + 8 * hf);
  t.q[1] = *(const v4u*)(row + kc + 16 + 8 * hf);
  return t.v;
}

// B-operand (32x16 bf16): lane holds col n = lane%16;
// VGPR0-3: K = 8*hf .. 8*hf+7 ; VGPR4-7: K = 16+8*hf .. 16+8*hf+7 (matches A).
static __device__ __forceinline__ v16bf load_b16(const unsigned short* row, int kc, int hf) {
  V16U t;
  const unsigned short* p = row + kc + 8 * hf;
  t.q[0] = *(const v4u*)(p);
  t.q[1] = *(const v4u*)(p + 16);
  return t.v;
}

static __device__ __forceinline__ v8f wmma_bf16(v16bf a, v16bf b, v8f c) {
  return __builtin_amdgcn_wmma_f32_16x16x32_bf16(false, a, false, b, (short)0, c, false, false);
}

// ---------------------------------------------------------------------------
// 32x64 output tile per wave, ping-pong software pipeline (k-unroll by 2).
// A:[M,K] bf16 row-major, BT:[N,K] bf16 row-major (pre-transposed weights).
// acc[mt][nt] covers rows mbase+mt*16.., cols nbase+nt*16..
// ---------------------------------------------------------------------------
static __device__ __forceinline__ void gemm32x64(const unsigned short* A,
                                                 const unsigned short* BT,
                                                 int K, int mbase, int nbase,
                                                 int lane, v8f acc[2][4]) {
  const int hf = lane >> 4, l16 = lane & 15;
  const unsigned short* arow0 = A + (size_t)(mbase + l16) * K;
  const unsigned short* arow1 = arow0 + (size_t)16 * K;
  const unsigned short* brow = BT + (size_t)(nbase + l16) * K;

  v16bf aA[2], bA[4], aB[2], bB[4];
  aA[0] = load_a16(arow0, 0, hf);
  aA[1] = load_a16(arow1, 0, hf);
#pragma unroll
  for (int nt = 0; nt < 4; ++nt) bA[nt] = load_b16(brow + (size_t)nt * 16 * K, 0, hf);

  for (int kc = 0; kc < K; kc += 64) {              // K % 64 == 0
    const int k1 = kc + 32;
    const int k2 = (kc + 64 < K) ? kc + 64 : 0;     // uniform; tail reloads 0
    __builtin_prefetch(arow0 + kc + 512, 0, 1);     // global_prefetch_b8
    __builtin_prefetch(arow1 + kc + 512, 0, 1);
    __builtin_prefetch(brow + kc + 512, 0, 1);

    // stage 0: load k1 into set B, compute on set A
    aB[0] = load_a16(arow0, k1, hf);
    aB[1] = load_a16(arow1, k1, hf);
#pragma unroll
    for (int nt = 0; nt < 4; ++nt) bB[nt] = load_b16(brow + (size_t)nt * 16 * K, k1, hf);
#pragma unroll
    for (int nt = 0; nt < 4; ++nt) {
      acc[0][nt] = wmma_bf16(aA[0], bA[nt], acc[0][nt]);
      acc[1][nt] = wmma_bf16(aA[1], bA[nt], acc[1][nt]);
    }
    // stage 1: load k2 into set A, compute on set B
    aA[0] = load_a16(arow0, k2, hf);
    aA[1] = load_a16(arow1, k2, hf);
#pragma unroll
    for (int nt = 0; nt < 4; ++nt) bA[nt] = load_b16(brow + (size_t)nt * 16 * K, k2, hf);
#pragma unroll
    for (int nt = 0; nt < 4; ++nt) {
      acc[0][nt] = wmma_bf16(aB[0], bB[nt], acc[0][nt]);
      acc[1][nt] = wmma_bf16(aB[1], bB[nt], acc[1][nt]);
    }
  }
}

// ------------------------------ conversions --------------------------------

__global__ __launch_bounds__(256) void cvt_bf16_kernel(const float* in,
                                                       unsigned short* out, int n) {
  int i = blockIdx.x * 256 + threadIdx.x;
  if (i < n) out[i] = f2bf(in[i]);
}

// in: [K, N] f32  ->  out: [N, K] bf16, LDS-tiled (coalesced both sides)
__global__ __launch_bounds__(256) void transpose_cvt_kernel(const float* in,
                                                            unsigned short* out,
                                                            int K, int N) {
  __shared__ unsigned short tile[32][33];
  const int tx = threadIdx.x & 31, ty = threadIdx.x >> 5;   // 32 x 8
  const int n0 = blockIdx.x * 32, k0 = blockIdx.y * 32;
#pragma unroll
  for (int j = 0; j < 4; ++j) {
    const int k = k0 + ty + j * 8;
    tile[ty + j * 8][tx] = f2bf(in[(size_t)k * N + n0 + tx]);
  }
  __syncthreads();
#pragma unroll
  for (int j = 0; j < 4; ++j) {
    const int n = n0 + ty + j * 8;
    out[(size_t)n * K + k0 + tx] = tile[tx][ty + j * 8];
  }
}

// ------------------------------ QKV GEMM -----------------------------------
// qkv[m, n] = x[m,:] @ Wqkv[:,n] + b[n];  n = which*1024 + h*64 + d
// Scatter: Q (scaled by SCALE) and K as [bh, t, d] bf16; V transposed [bh, d, t].

__global__ __launch_bounds__(256) void gemm_qkv_kernel(const unsigned short* xb,
                                                       const unsigned short* WT,
                                                       const float* bias,
                                                       unsigned short* Q,
                                                       unsigned short* Kd,
                                                       unsigned short* Vt) {
  const int tid = threadIdx.x, wave = tid >> 5, lane = tid & 31;
  const int hf = lane >> 4, l16 = lane & 15;
  const int gw = blockIdx.x * 8 + wave;
  const int mtile = gw / 48, ncol = gw - mtile * 48;   // 256 mtiles x 48 ncols
  const int mbase = mtile * 32, nbase = ncol * 64;

  v8f acc[2][4] = {};
  gemm32x64(xb, WT, C_, mbase, nbase, lane, acc);

  const int which = nbase >> 10;
  const int h = (nbase & 1023) >> 6;
  const int b = mbase >> 11;          // / T_
  const int t0 = mbase & (T_ - 1);
  const int bh = b * H_ + h;

#pragma unroll
  for (int mt = 0; mt < 2; ++mt) {
#pragma unroll
    for (int nt = 0; nt < 4; ++nt) {
      const int d = nt * 16 + l16;
      const float bs = bias[nbase + nt * 16 + l16];
#pragma unroll
      for (int r = 0; r < 8; ++r) {
        const int t = t0 + mt * 16 + r + 8 * hf;
        const float v = acc[mt][nt][r] + bs;
        if (which == 0)      Q [((size_t)(bh * T_ + t)) * D_ + d] = f2bf(v * SCALE_);
        else if (which == 1) Kd[((size_t)(bh * T_ + t)) * D_ + d] = f2bf(v);
        else                 Vt[((size_t)(bh * D_ + d)) * T_ + t] = f2bf(v);
      }
    }
  }
}

// --------------------------- flash attention -------------------------------
// One wave per 16-row q tile; 32-key blocks; online softmax in f32.
// K-tile operands ping-pong across paired blocks (no rotation copies);
// V-tile loads hoisted above the softmax so ~60 VALU/trans ops hide them.
// P transposed C-layout -> A-layout via 1KB/wave LDS + wave-local dscnt fence.

static __device__ __forceinline__ void load_ktile(const unsigned short* kbp,
                                                  int kb, int hf, int l16,
                                                  v16bf bk[4]) {
#pragma unroll
  for (int st = 0; st < 2; ++st) {
    const int key = kb + st * 16 + l16;
    const int kcl = key < T_ ? key : T_ - 1;            // clamp (masked anyway)
    const unsigned short* krow = kbp + (size_t)kcl * D_;
    bk[2 * st]     = load_b16(krow, 0, hf);
    bk[2 * st + 1] = load_b16(krow, 32, hf);
  }
}

__global__ __launch_bounds__(256) void fa_kernel(const unsigned short* Q,
                                                 const unsigned short* Kd,
                                                 const unsigned short* Vt,
                                                 const signed char* kpad,
                                                 unsigned short* O) {
  __shared__ unsigned short lds[8 * 512];
  const int tid = threadIdx.x, wave = tid >> 5, lane = tid & 31;
  const int hf = lane >> 4, l16 = lane & 15;
  const int gw = blockIdx.x * 8 + wave;
  const int bh = gw >> 7;            // / (T/16)
  const int qi = gw & 127;
  const int qbase = qi << 4;
  const int b = bh >> 4;             // / H
  const int h = bh & 15;
  unsigned short* pbuf = lds + wave * 512;   // 16 rows x 32 keys bf16

  const unsigned short* qrow = Q + (size_t)(bh * T_ + qbase + l16) * D_;
  const v16bf a0 = load_a16(qrow, 0, hf);    // d 0..31  (SCALE pre-folded)
  const v16bf a1 = load_a16(qrow, 32, hf);   // d 32..63
  const unsigned short* kbp = Kd + (size_t)bh * T_ * D_;
  const unsigned short* vbp = Vt + (size_t)bh * D_ * T_;
  const signed char* padrow = kpad + (size_t)b * T_;

  v8f acc[4] = {};
  float mrow[8], lrow[8];
#pragma unroll
  for (int r = 0; r < 8; ++r) { mrow[r] = -3.0e38f; lrow[r] = 0.0f; }

  // One 32-key block: prefetch next K tile into bkn, compute on bk.
  auto fa_step = [&](int kb, v16bf* bk, v16bf* bkn, int kbNext) {
    load_ktile(kbp, kbNext, hf, l16, bkn);

    // ---- S = Q K^T (two 16-key tiles) ----
    v8f s[2] = {};
    s[0] = wmma_bf16(a0, bk[0], s[0]);
    s[0] = wmma_bf16(a1, bk[1], s[0]);
    s[1] = wmma_bf16(a0, bk[2], s[1]);
    s[1] = wmma_bf16(a1, bk[3], s[1]);

    // ---- hoist V-tile loads (independent of softmax) ----
    v16bf bv[4];
#pragma unroll
    for (int nt = 0; nt < 4; ++nt) {
      const int d = nt * 16 + l16;
      bv[nt] = load_b16(vbp + (size_t)d * T_ + kb, 0, hf);
    }

    // ---- masks + online softmax ----
    const int key0 = kb + l16, key1 = kb + 16 + l16;
    const int pad0 = padrow[key0 < T_ ? key0 : T_ - 1];
    const int pad1 = padrow[key1 < T_ ? key1 : T_ - 1];
    float p0[8], p1[8];
#pragma unroll
    for (int r = 0; r < 8; ++r) {
      const int row = qbase + r + 8 * hf;
      float s0 = s[0][r], s1 = s[1][r];
      if (key0 > row || pad0) s0 = -3.0e38f;
      if (key1 > row || pad1) s1 = -3.0e38f;
      float m = fmaxf(s0, s1);
      m = fmaxf(m, __shfl_xor(m, 1, 32));
      m = fmaxf(m, __shfl_xor(m, 2, 32));
      m = fmaxf(m, __shfl_xor(m, 4, 32));
      m = fmaxf(m, __shfl_xor(m, 8, 32));
      const float mnew = fmaxf(mrow[r], m);
      const float corr = __expf(mrow[r] - mnew);
      const float e0 = __expf(s0 - mnew);
      const float e1 = __expf(s1 - mnew);
      float rs = e0 + e1;
      rs += __shfl_xor(rs, 1, 32);
      rs += __shfl_xor(rs, 2, 32);
      rs += __shfl_xor(rs, 4, 32);
      rs += __shfl_xor(rs, 8, 32);
      lrow[r] = lrow[r] * corr + rs;
      mrow[r] = mnew;
      p0[r] = e0; p1[r] = e1;
#pragma unroll
      for (int nt = 0; nt < 4; ++nt) acc[nt][r] *= corr;
    }

    // ---- transpose P (C-layout -> A-layout) through LDS ----
#pragma unroll
    for (int r = 0; r < 8; ++r) {
      const int row = r + 8 * hf;
      pbuf[row * 32 + l16]      = f2bf(p0[r]);
      pbuf[row * 32 + 16 + l16] = f2bf(p1[r]);
    }
    asm volatile("s_wait_dscnt 0x0" ::: "memory");   // wave-local LDS RAW fence
    const v16bf pa = load_a16(pbuf + (size_t)l16 * 32, 0, hf);

    // ---- acc += P @ V ----
#pragma unroll
    for (int nt = 0; nt < 4; ++nt) acc[nt] = wmma_bf16(pa, bv[nt], acc[nt]);
  };

  const int NB = (qbase + 47) / 32;          // # of 32-key blocks (>= 1)
  v16bf bufA[4], bufB[4];
  load_ktile(kbp, 0, hf, l16, bufA);
  int i = 0;
  for (; i + 2 <= NB; i += 2) {              // paired ping-pong blocks
    fa_step(i * 32, bufA, bufB, (i + 1) * 32);
    fa_step((i + 1) * 32, bufB, bufA, (i + 2 < NB) ? (i + 2) * 32 : 0);
  }
  if (i < NB)                                // odd tail block
    fa_step(i * 32, bufA, bufB, 0);

  // ---- normalize + store bf16 [b, t, h*64 + d] ----
#pragma unroll
  for (int nt = 0; nt < 4; ++nt) {
    const int d = h * 64 + nt * 16 + l16;
#pragma unroll
    for (int r = 0; r < 8; ++r) {
      const int t = qbase + r + 8 * hf;
      O[(size_t)(b * T_ + t) * C_ + d] = f2bf(acc[nt][r] / lrow[r]);
    }
  }
}

// ------------------------------ proj GEMM ----------------------------------

__global__ __launch_bounds__(256) void gemm_proj_kernel(const unsigned short* A,
                                                        const unsigned short* WT,
                                                        const float* bias,
                                                        float* out) {
  const int tid = threadIdx.x, wave = tid >> 5, lane = tid & 31;
  const int hf = lane >> 4, l16 = lane & 15;
  const int gw = blockIdx.x * 8 + wave;
  const int mtile = gw >> 4, ncol = gw & 15;           // 256 mtiles x 16 ncols
  const int mbase = mtile * 32, nbase = ncol * 64;

  v8f acc[2][4] = {};
  gemm32x64(A, WT, C_, mbase, nbase, lane, acc);

#pragma unroll
  for (int mt = 0; mt < 2; ++mt) {
#pragma unroll
    for (int nt = 0; nt < 4; ++nt) {
      const int n = nbase + nt * 16 + l16;
      const float bs = bias[n];
#pragma unroll
      for (int r = 0; r < 8; ++r)
        out[(size_t)(mbase + mt * 16 + r + 8 * hf) * C_ + n] = acc[mt][nt][r] + bs;
    }
  }
}

// ------------------------------ launcher -----------------------------------

extern "C" void kernel_launch(void* const* d_in, const int* in_sizes, int n_in,
                              void* d_out, int out_size, void* d_ws, size_t ws_size,
                              hipStream_t stream) {
  const float*       x     = (const float*)d_in[0];
  const signed char* kpad  = (const signed char*)d_in[1];
  const float*       Wqkv  = (const float*)d_in[2];
  const float*       bqkv  = (const float*)d_in[3];
  const float*       Wproj = (const float*)d_in[4];
  const float*       bproj = (const float*)d_in[5];
  float*             out   = (float*)d_out;

  const int M = B_ * T_;                       // 8192
  char* ws = (char*)d_ws;                      // bf16 scratch, ~92 MB total
  unsigned short* xb     = (unsigned short*)(ws);
  unsigned short* WqkvT  = (unsigned short*)(ws + (size_t)16777216);
  unsigned short* WprojT = (unsigned short*)(ws + (size_t)23068672);
  unsigned short* Qb     = (unsigned short*)(ws + (size_t)25165824);
  unsigned short* Kb     = (unsigned short*)(ws + (size_t)41943040);
  unsigned short* Vtb    = (unsigned short*)(ws + (size_t)58720256);
  unsigned short* AOb    = (unsigned short*)(ws + (size_t)75497472);

  cvt_bf16_kernel<<<(M * C_ + 255) / 256, 256, 0, stream>>>(x, xb, M * C_);
  transpose_cvt_kernel<<<dim3((3 * C_) / 32, C_ / 32), 256, 0, stream>>>(Wqkv, WqkvT, C_, 3 * C_);
  transpose_cvt_kernel<<<dim3(C_ / 32, C_ / 32), 256, 0, stream>>>(Wproj, WprojT, C_, C_);

  gemm_qkv_kernel<<<(M / 32) * (3 * C_ / 64) / 8, 256, 0, stream>>>(xb, WqkvT, bqkv, Qb, Kb, Vtb);
  fa_kernel<<<(B_ * H_ * (T_ / 16)) / 8, 256, 0, stream>>>(Qb, Kb, Vtb, kpad, AOb);
  gemm_proj_kernel<<<(M / 32) * (C_ / 64) / 8, 256, 0, stream>>>(AOb, WprojT, bproj, out);
}
